// Projection_25237227832002
// MI455X (gfx1250) — matrix operand order repaired and checked by reading, not compiled, not run on hardware
//
#include <hip/hip_runtime.h>
#include <stdint.h>

// ---------------------------------------------------------------------------
// emb = (mask @ mlp_out) / max(counts,1),  mask = (M != 0), M[50000,2000] f32
// Masked GEMM M=50000,N=128,K=2000 on v_wmma_f32_16x16x32_bf16 (wave32).
//  - B (bf16 mlp table) staged per-block in LDS via global_load_async_to_lds
//    (double buffered on ASYNCcnt, ONE barrier per iteration)
//  - counts = mask @ ones on a 9th WMMA accumulator
//  - mask weight w = 2^-126 (bf16 0x0080): pack = v_perm + v_pk_min_u16 only;
//    w cancels in (w*mask@B)/(w*count), epilogue clamps with w
//  - A raw data pipelined TWO chunks ahead (two persistent buffers, static
//    even/odd selection) so no WMMA operand aliases a pending global load
// ---------------------------------------------------------------------------
#define NUM_L   50000
#define NUM_S   2000
#define SPAD    2048
#define EMB     128
#define NTILES  3125
#define NCHUNK  63            // ceil(2000/32); last chunk half-valid
#define WAVES_PER_BLOCK 8
#define CHUNK_ELTS (EMB * 32) // 8 KB per staged K-chunk
#define WCLAMP  1.17549435e-38f   // 2^-126 == mask weight w

typedef __bf16          v16bf __attribute__((ext_vector_type(16)));
typedef float           v8f   __attribute__((ext_vector_type(8)));
typedef unsigned        u32x4 __attribute__((ext_vector_type(4)));

// ---------------------------------------------------------------------------
// Stage 1: mlp table transposed: mlpT[d][j] bf16, j padded to 2048 with zeros
// ---------------------------------------------------------------------------
__device__ __forceinline__ unsigned short f32_to_bf16_rne(float f) {
    unsigned u = __float_as_uint(f);
    u += 0x7FFFu + ((u >> 16) & 1u);
    return (unsigned short)(u >> 16);
}

__global__ __launch_bounds__(256)
void mlp_table_kernel(const float* __restrict__ W1, const float* __restrict__ b1,
                      const float* __restrict__ W2, const float* __restrict__ b2,
                      unsigned short* __restrict__ mlpT) {
    const int t = blockIdx.x * blockDim.x + threadIdx.x;
    const int j = t & (SPAD - 1);
    const int d = t >> 11;
    float acc = 0.0f;
    if (j < NUM_S) {
        acc = b2[d];
        const float jf = (float)j;
#pragma unroll
        for (int k = 0; k < 16; ++k) {
            const float h = fmaxf(fmaf(jf, W1[k], b1[k]), 0.0f);
            acc = fmaf(h, W2[k * EMB + d], acc);
        }
    }
    mlpT[(size_t)d * SPAD + j] = f32_to_bf16_rne(acc);
}

// ---------------------------------------------------------------------------
// Stage 2 helpers
// ---------------------------------------------------------------------------
__device__ __forceinline__ u32x4 ld_nt4(const unsigned* p) {
    return __builtin_nontemporal_load(reinterpret_cast<const u32x4*>(p));
}

// two f32 bit patterns -> packed bf16 {w or 0, w or 0}, w = 2^-126 (0x0080).
// hi16 of any nonzero normal f32 is >= 0x0080, so pk_min clamps it to w.
__device__ __forceinline__ unsigned pack2(unsigned u0, unsigned u1, unsigned cmin) {
    const unsigned t = __builtin_amdgcn_perm(u1, u0, 0x07060302); // {hi16 u1, hi16 u0}
    unsigned m;
    asm("v_pk_min_u16 %0, %1, %2" : "=v"(m) : "v"(t), "v"(cmin));
    return m;
}

struct Raw4 { u32x4 a, b, c, d; };
union  AU   { unsigned u[8]; v16bf v; };
union  BU   { u32x4 q[2]; v16bf v; };

__device__ __forceinline__ Raw4 load_raw(const unsigned* __restrict__ Mrow,
                                         int k0, int sel) {
    Raw4 r;
    const int k1 = k0 + sel * 8;
    const unsigned* p = Mrow + k1;
    r.a = ld_nt4(p);
    r.b = ld_nt4(p + 4);
    if (k1 + 24 <= NUM_S) {            // g2,g3 cover floats [k1+16, k1+24)
        r.c = ld_nt4(p + 16);
        r.d = ld_nt4(p + 20);
    } else {
        r.c = (u32x4){0, 0, 0, 0};     // K >= 2000 -> mask 0
        r.d = (u32x4){0, 0, 0, 0};
    }
    return r;
}

__device__ __forceinline__ void pack_A(const Raw4& r, unsigned cmin, AU& A) {
    A.u[0] = pack2(r.a.x, r.a.y, cmin);
    A.u[1] = pack2(r.a.z, r.a.w, cmin);
    A.u[2] = pack2(r.b.x, r.b.y, cmin);
    A.u[3] = pack2(r.b.z, r.b.w, cmin);
    A.u[4] = pack2(r.c.x, r.c.y, cmin);
    A.u[5] = pack2(r.c.z, r.c.w, cmin);
    A.u[6] = pack2(r.d.x, r.d.y, cmin);
    A.u[7] = pack2(r.d.z, r.d.w, cmin);
}

// CDNA5 async global->LDS copy (GVS mode: saddr + voffset), 16B per lane
__device__ __forceinline__ void async_copy_b128(unsigned lds_off, unsigned gl_off,
                                                uint64_t sbase) {
    asm volatile("global_load_async_to_lds_b128 %0, %1, %2"
                 :
                 : "v"(lds_off), "v"(gl_off), "s"(sbase)
                 : "memory");
}

// Stage one 8 KB B chunk into LDS: thread t copies dim d=t>>1, K-half=t&1
__device__ __forceinline__ void issue_chunk(uint64_t mlpT, int k0,
                                            unsigned short* buf, int t) {
    const int d    = t >> 1;
    const int half = t & 1;
    const unsigned goff = (unsigned)(((size_t)d * SPAD + k0 + half * 16) * 2);
    const unsigned loff = (unsigned)(uintptr_t)(buf + (d * 32 + half * 16));
    async_copy_b128(loff,      goff,      mlpT);
    async_copy_b128(loff + 16, goff + 16, mlpT);
}

// ds-load all 8 B tiles, then run the 9 WMMAs (A already packed in regs)
__device__ __forceinline__ void compute_step(const unsigned short* __restrict__ ldsB,
                                             const AU& A, int sel, int rsub,
                                             v8f (&acc)[8], v8f& cacc,
                                             const v16bf& ones) {
    BU B[8];
#pragma unroll
    for (int nt = 0; nt < 8; ++nt) {
        const unsigned short* bp = ldsB + (nt * 16 + rsub) * 32 + sel * 8;
        B[nt].q[0] = *reinterpret_cast<const u32x4*>(bp);
        B[nt].q[1] = *reinterpret_cast<const u32x4*>(bp + 16);
    }
    __builtin_amdgcn_sched_barrier(0);   // all ds loads issue before WMMAs

    cacc = __builtin_amdgcn_wmma_f32_16x16x32_bf16(     // A-only: runs first
        false, A.v, false, ones, (short)0, cacc, false, false);
#pragma unroll
    for (int nt = 0; nt < 8; ++nt)
        acc[nt] = __builtin_amdgcn_wmma_f32_16x16x32_bf16(
            false, A.v, false, B[nt].v, (short)0, acc[nt], false, false);
}

// One pipelined iteration. rb holds this chunk's raw A data (loaded 2 chunks
// ago); it is consumed by pack_A and then overwritten with chunk kk+2.
__device__ __forceinline__ void iter_body(int kk, bool nextB, bool nextRaw,
                                          Raw4& rb,
                                          const unsigned short* ldsCur,
                                          unsigned short* ldsNext,
                                          const unsigned* __restrict__ Mrow,
                                          uint64_t mlpT_u, int t, int sel, int rsub,
                                          unsigned cmin, const v16bf& ones,
                                          v8f (&acc)[8], v8f& cacc) {
    AU A;
    pack_A(rb, cmin, A);                               // overlaps barrier wait
    asm volatile("s_wait_asynccnt 0x0" ::: "memory");  // my B writes done
    __syncthreads();        // all writes done AND all reads of ldsNext done
    if (nextB)   issue_chunk(mlpT_u, (kk + 1) * 32, ldsNext, t);
    if (nextRaw) rb = load_raw(Mrow, (kk + 2) * 32, sel);   // 2 chunks ahead
    __builtin_prefetch(Mrow + kk * 32 + 512, 0, 3);
    compute_step(ldsCur, A, sel, rsub, acc, cacc, ones);
}

__global__ __launch_bounds__(256)
void masked_gemm_kernel(const float* __restrict__ M,
                        const unsigned short* __restrict__ mlpT,
                        float* __restrict__ out) {
    __shared__ __align__(16) unsigned short lds_b[2][CHUNK_ELTS];   // 16 KB

    const int t    = threadIdx.x;
    const int lane = t & 31;
    const int wave = t >> 5;
    int tile = blockIdx.x * WAVES_PER_BLOCK + wave;      // wave-uniform
    if (tile >= NTILES) tile = NTILES - 1;               // dup waves store
                                                         // identical values
    const int sel  = lane >> 4;
    const int rsub = lane & 15;
    const unsigned* Mrow =
        reinterpret_cast<const unsigned*>(M) + (size_t)(tile * 16 + rsub) * NUM_S;
    const uint64_t mlpT_u = (uint64_t)(uintptr_t)mlpT;
    const unsigned cmin = 0x00800080u;                   // packed bf16 w,w

    union { unsigned u[8]; v16bf v; } ONES;
#pragma unroll
    for (int i = 0; i < 8; ++i) ONES.u[i] = 0x3F803F80u; // bf16 1.0 pairs

    const v8f vzero = {0.f, 0.f, 0.f, 0.f, 0.f, 0.f, 0.f, 0.f};
    v8f acc[8];
#pragma unroll
    for (int i = 0; i < 8; ++i) acc[i] = vzero;
    v8f cacc = vzero;

    // Prologue: B chunk 0 async -> LDS buf0; A raw chunks 0 and 1 -> regs
    issue_chunk(mlpT_u, 0, lds_b[0], t);
    Raw4 rbuf0 = load_raw(Mrow, 0,  sel);   // even chunks
    Raw4 rbuf1 = load_raw(Mrow, 32, sel);   // odd chunks

    for (int kk = 0; kk < 60; kk += 2) {
        iter_body(kk,     true, true, rbuf0, lds_b[0], lds_b[1], Mrow, mlpT_u,
                  t, sel, rsub, cmin, ONES.v, acc, cacc);
        iter_body(kk + 1, true, true, rbuf1, lds_b[1], lds_b[0], Mrow, mlpT_u,
                  t, sel, rsub, cmin, ONES.v, acc, cacc);
    }
    iter_body(60, true,  true,  rbuf0, lds_b[0], lds_b[1], Mrow, mlpT_u,
              t, sel, rsub, cmin, ONES.v, acc, cacc);   // loads chunk 62
    iter_body(61, true,  false, rbuf1, lds_b[1], lds_b[0], Mrow, mlpT_u,
              t, sel, rsub, cmin, ONES.v, acc, cacc);   // issues B chunk 62
    iter_body(62, false, false, rbuf0, lds_b[0], lds_b[1], Mrow, mlpT_u,
              t, sel, rsub, cmin, ONES.v, acc, cacc);   // K tail, A upper = 0

    // Epilogue: D layout: VGPR v -> row v (lanes 0-15) / v+8 (lanes 16-31),
    // col = nt*16 + rsub. cacc[v] = w * count; w cancels in the ratio.
    const int obase = tile * 16;
#pragma unroll
    for (int v = 0; v < 8; ++v) {
        const float inv = 1.0f / fmaxf(cacc[v], WCLAMP);
        float* orow = out + (size_t)(obase + v + sel * 8) * EMB + rsub;
#pragma unroll
        for (int nt = 0; nt < 8; ++nt) orow[nt * 16] = acc[nt][v] * inv;
    }
}

// ---------------------------------------------------------------------------
// Launch
// ---------------------------------------------------------------------------
extern "C" void kernel_launch(void* const* d_in, const int* in_sizes, int n_in,
                              void* d_out, int out_size, void* d_ws, size_t ws_size,
                              hipStream_t stream) {
    const float* M  = (const float*)d_in[0];   // [50000, 2000]
    const float* W1 = (const float*)d_in[1];   // [1, 16]
    const float* b1 = (const float*)d_in[2];   // [16]
    const float* W2 = (const float*)d_in[3];   // [16, 128]
    const float* b2 = (const float*)d_in[4];   // [128]
    float* out = (float*)d_out;                // [50000, 128]

    unsigned short* mlpT = (unsigned short*)d_ws;   // [128][2048] bf16, 512 KB

    mlp_table_kernel<<<(SPAD * EMB) / 256, 256, 0, stream>>>(W1, b1, W2, b2, mlpT);

    const int blocks = (NTILES + WAVES_PER_BLOCK - 1) / WAVES_PER_BLOCK;  // 391
    masked_gemm_kernel<<<blocks, 256, 0, stream>>>(M, mlpT, out);
}
